// aggregator_14345190769249
// MI455X (gfx1250) — matrix operand order repaired and checked by reading, not compiled, not run on hardware
//
#include <hip/hip_runtime.h>
#include <hip/hip_bf16.h>

// ---------------------------------------------------------------------------
// CDNA5 (gfx1250) implementation, wave32 + v_wmma_f32_16x16x32_bf16.
//
// Pipeline:
//   memset   : zero embed_matrix region of d_out
//   proj     : i_proj  = i_weight @ iW + ib           (ws)       [WMMA]
//   proj     : out_fea = u_weight[nodes] @ uW + ub    (d_out)    [WMMA]
//   attn     : per-node 2-layer MLP + softmax + weighted gather  [WMMA]
//
// i_proj (51 MB) stays resident in the 192 MB L2, so the 614 MB neighbor
// gather is L2 traffic, not HBM.  All matmuls are bf16 A/B, f32 accumulate.
// ---------------------------------------------------------------------------

typedef __bf16     v16bf __attribute__((ext_vector_type(16)));
typedef float      v8f   __attribute__((ext_vector_type(8)));
typedef float      f32x4 __attribute__((ext_vector_type(4)));
typedef float      f32x2 __attribute__((ext_vector_type(2)));
typedef unsigned   u32x4 __attribute__((ext_vector_type(4)));

#define EDIM 64
#define FDIM 128
#define KNEI 24

__device__ __forceinline__ v8f wmma_bf16(v16bf a, v16bf b, v8f c) {
    // D = A(16x32 bf16) x B(32x16 bf16) + C(16x16 f32)
    return __builtin_amdgcn_wmma_f32_16x16x32_bf16(
        /*neg_a=*/false, a, /*neg_b=*/false, b,
        /*c_mod=*/(short)0, c, /*reuse_a=*/false, /*reuse_b=*/false);
}

// Build A-operand register per ISA 16-bit A layout. Caller passes the two
// contiguous 8-element chunks for this lane (lo = K base, hi = K base+16).
__device__ __forceinline__ v16bf pack16f(const float* lo, const float* hi) {
    f32x4 a = *(const f32x4*)lo;
    f32x4 b = *(const f32x4*)(lo + 4);
    f32x4 c = *(const f32x4*)hi;
    f32x4 d = *(const f32x4*)(hi + 4);
    v16bf r;
#pragma unroll
    for (int i = 0; i < 4; ++i) {
        r[i]      = (__bf16)a[i];
        r[4 + i]  = (__bf16)b[i];
        r[8 + i]  = (__bf16)c[i];
        r[12 + i] = (__bf16)d[i];
    }
    return r;
}

__device__ __forceinline__ v16bf pack16h(const __bf16* lo, const __bf16* hi) {
    union { v16bf v; u32x4 q[2]; } u;
    u.q[0] = *(const u32x4*)lo;   // ds_load_b128
    u.q[1] = *(const u32x4*)hi;   // ds_load_b128
    return u.v;
}

// ---------------------------------------------------------------------------
// proj_kernel: out[r,:] = feat[gather?gather[r]:r, 0:128] @ W(128x64) + bias
// One wave computes 32 output rows (2 M-tiles x 4 N-tiles x 4 K-tiles).
// nrows is a multiple of 32 (100000, 200000), so waves exit uniformly.
// ---------------------------------------------------------------------------
__global__ void __launch_bounds__(128)
proj_kernel(const float* __restrict__ feat,
            const int*   __restrict__ gather,
            const float* __restrict__ W,
            const float* __restrict__ bias,
            float*       __restrict__ out,
            int nrows)
{
    // B tiles pre-swizzled to WMMA B-operand lane layout:
    // tile(t,j): lane holds column j*16+(lane&15), K halfs in table order (8h+s, 16+8h+s-8).
    __shared__ __align__(32) __bf16 Bt[16][32][16];   // 16 KB

    const int tid = threadIdx.x;
    for (int f = tid; f < 16 * 512; f += 128) {
        int tile = f >> 9, rem = f & 511;
        int l = rem >> 4, h = rem & 15;
        int t = tile >> 2, j = tile & 3;
        int k = t * 32 + ((l < 16) ? 0 : 8) + (h & 7) + ((h & 8) << 1);
        int c = j * 16 + (l & 15);
        Bt[tile][l][h] = (__bf16)W[k * EDIM + c];
    }
    __syncthreads();

    const int  lane  = tid & 31;
    const long rbase = (long)(blockIdx.x * 4 + (tid >> 5)) * 32;
    if (rbase >= nrows) return;                 // uniform per wave

    const int mrow  = lane & 15;
    const int chunk = (lane < 16) ? 0 : 8;      // A-layout K sub-chunk

    v16bf a[2][4];
#pragma unroll
    for (int m = 0; m < 2; ++m) {
        long row = rbase + m * 16 + mrow;
        long src = gather ? (long)gather[row] : row;
        const float* rp = feat + src * FDIM;
#pragma unroll
        for (int t = 0; t < 4; ++t) {
            const float* p = rp + t * 32 + chunk;
            a[m][t] = pack16f(p, p + 16);
        }
    }

    const int colb   = lane & 15;
    const int rowoff = (lane < 16) ? 0 : 8;     // C/D layout M offset
#pragma unroll
    for (int j = 0; j < 4; ++j) {
        float bv = bias[j * 16 + colb];
        v8f acc0 = {bv, bv, bv, bv, bv, bv, bv, bv};
        v8f acc1 = acc0;
#pragma unroll
        for (int t = 0; t < 4; ++t) {
            v16bf b = *(const v16bf*)&Bt[t * 4 + j][lane][0];
            acc0 = wmma_bf16(a[0][t], b, acc0);
            acc1 = wmma_bf16(a[1][t], b, acc1);
        }
#pragma unroll
        for (int v = 0; v < 8; ++v) {
            long r0 = rbase + v + rowoff;
            long r1 = rbase + 16 + v + rowoff;
            int  c  = j * 16 + colb;
            out[r0 * EDIM + c] = acc0[v];
            out[r1 * EDIM + c] = acc1[v];
        }
    }
}

// ---------------------------------------------------------------------------
// attn_kernel: one wave = one node. K=24 neighbors padded to M=32.
// Layer1: [32,128]@[128,64]  (K-tiles 2,3 broadcast q  == concat trick)
// Layer2: [32,64]@[64,64]
// Then scores -> softmax -> attention-weighted sum -> scatter.
// ---------------------------------------------------------------------------
__global__ void __launch_bounds__(64)
attn_kernel(const int*   __restrict__ nodes,
            const float* __restrict__ nfeat,
            const int*   __restrict__ neigh,
            const float* __restrict__ aW1, const float* __restrict__ ab1,
            const float* __restrict__ aW2, const float* __restrict__ ab2,
            const float* __restrict__ aW3, const float* __restrict__ ab3,
            const float* __restrict__ iproj,
            const float* __restrict__ nodes_fea,
            float*       __restrict__ emb_out,
            int N)
{
    __shared__ __align__(32) __bf16 W1t[16][32][16];   // 16 KB  (K=128)
    __shared__ __align__(32) __bf16 W2t[8][32][16];    //  8 KB  (K=64)
    __shared__ float  w3s[64];
    __shared__ __align__(16) float  neiS[2][32][68];   // f32 neighbors (padded)
    __shared__ __align__(16) __bf16 hS[2][32][80];     // layer transition
    __shared__ __align__(16) float  qS[2][64];
    __shared__ float  scS[2][32];
    __shared__ int    idxS[2][32];

    const int tid = threadIdx.x;
    for (int f = tid; f < 16 * 512; f += 64) {
        int tile = f >> 9, rem = f & 511;
        int l = rem >> 4, h = rem & 15;
        int t = tile >> 2, j = tile & 3;
        W1t[tile][l][h] =
            (__bf16)aW1[(t * 32 + ((l < 16) ? 0 : 8) + (h & 7) + ((h & 8) << 1)) * EDIM + j * 16 + (l & 15)];
    }
    for (int f = tid; f < 8 * 512; f += 64) {
        int tile = f >> 9, rem = f & 511;
        int l = rem >> 4, h = rem & 15;
        int t = tile >> 2, j = tile & 3;
        W2t[tile][l][h] =
            (__bf16)aW2[(t * 32 + ((l < 16) ? 0 : 8) + (h & 7) + ((h & 8) << 1)) * EDIM + j * 16 + (l & 15)];
    }
    if (tid < 64) w3s[tid] = aW3[tid];
    __syncthreads();

    const int lane = tid & 31, w = tid >> 5;
    int node = blockIdx.x * 2 + w;
    if (node >= N) node = N - 1;     // clamp: duplicate waves write identical data

    // neighbor indices (pad rows 24..31 with idx[0]; their output is ignored)
    idxS[w][lane] = neigh[(long)node * KNEI + ((lane < KNEI) ? lane : 0)];

    // q = nodes_fea[node] + n_feature[node]
    {
        f32x2 a = *(const f32x2*)&nodes_fea[(long)node * EDIM + lane * 2];
        f32x2 b = *(const f32x2*)&nfeat[(long)node * EDIM + lane * 2];
        *(f32x2*)&qS[w][lane * 2] = a + b;
    }
    __syncthreads();

    // gather neighbor rows from i_proj (L2-resident) into LDS, coalesced
#pragma unroll 4
    for (int r = 0; r < 32; ++r) {
        long src = (long)idxS[w][r] * EDIM;
        *(f32x2*)&neiS[w][r][lane * 2] = *(const f32x2*)&iproj[src + lane * 2];
    }
    __syncthreads();

    const int mrow  = lane & 15;
    const int chunk = (lane < 16) ? 0 : 8;
    const int colb  = lane & 15;
    const int rowoff = (lane < 16) ? 0 : 8;

    // A operands: neighbor part (K 0..63) + q broadcast part (K 64..127)
    v16bf a1[2][2], aq[2];
#pragma unroll
    for (int m = 0; m < 2; ++m)
#pragma unroll
        for (int t = 0; t < 2; ++t) {
            const float* p = &neiS[w][m * 16 + mrow][t * 32 + chunk];
            a1[m][t] = pack16f(p, p + 16);
        }
#pragma unroll
    for (int t = 0; t < 2; ++t) {
        const float* p = &qS[w][t * 32 + chunk];
        aq[t] = pack16f(p, p + 16);
    }

    // -------- layer 1: h1 = relu([neigh|q] @ aW1 + ab1) --------
#pragma unroll
    for (int j = 0; j < 4; ++j) {
        float bv = ab1[j * 16 + colb];
        v8f acc0 = {bv, bv, bv, bv, bv, bv, bv, bv};
        v8f acc1 = acc0;
#pragma unroll
        for (int t = 0; t < 4; ++t) {
            v16bf b = *(const v16bf*)&W1t[t * 4 + j][lane][0];
            v16bf am0 = (t < 2) ? a1[0][t] : aq[t - 2];
            v16bf am1 = (t < 2) ? a1[1][t] : aq[t - 2];
            acc0 = wmma_bf16(am0, b, acc0);
            acc1 = wmma_bf16(am1, b, acc1);
        }
#pragma unroll
        for (int v = 0; v < 8; ++v) {
            float x0 = acc0[v] > 0.f ? acc0[v] : 0.f;
            float x1 = acc1[v] > 0.f ? acc1[v] : 0.f;
            hS[w][v + rowoff][j * 16 + colb]      = (__bf16)x0;
            hS[w][16 + v + rowoff][j * 16 + colb] = (__bf16)x1;
        }
    }
    __syncthreads();

    // -------- layer 2: h2 = relu(h1 @ aW2 + ab2) --------
    v16bf a2[2][2];
#pragma unroll
    for (int m = 0; m < 2; ++m)
#pragma unroll
        for (int t = 0; t < 2; ++t) {
            const __bf16* p = &hS[w][m * 16 + mrow][t * 32 + chunk];
            a2[m][t] = pack16h(p, p + 16);
        }
    __syncthreads();   // all A2 reads done before hS is overwritten

#pragma unroll
    for (int j = 0; j < 4; ++j) {
        float bv = ab2[j * 16 + colb];
        v8f acc0 = {bv, bv, bv, bv, bv, bv, bv, bv};
        v8f acc1 = acc0;
#pragma unroll
        for (int t = 0; t < 2; ++t) {
            v16bf b = *(const v16bf*)&W2t[t * 4 + j][lane][0];
            acc0 = wmma_bf16(a2[0][t], b, acc0);
            acc1 = wmma_bf16(a2[1][t], b, acc1);
        }
#pragma unroll
        for (int v = 0; v < 8; ++v) {
            float x0 = acc0[v] > 0.f ? acc0[v] : 0.f;
            float x1 = acc1[v] > 0.f ? acc1[v] : 0.f;
            hS[w][v + rowoff][j * 16 + colb]      = (__bf16)x0;
            hS[w][16 + v + rowoff][j * 16 + colb] = (__bf16)x1;
        }
    }
    __syncthreads();

    // -------- scores + softmax --------
    if (lane < KNEI) {
        float s = ab3[0];
        const __bf16* hr = &hS[w][lane][0];
#pragma unroll
        for (int c = 0; c < EDIM; ++c) s += (float)hr[c] * w3s[c];
        scS[w][lane] = s;
    } else {
        scS[w][lane] = -1e30f;
    }
    __syncthreads();

    float mx = -1e30f;
#pragma unroll
    for (int k = 0; k < KNEI; ++k) mx = fmaxf(mx, scS[w][k]);
    float e[KNEI], sum = 0.f;
#pragma unroll
    for (int k = 0; k < KNEI; ++k) { e[k] = __expf(scS[w][k] - mx); sum += e[k]; }
    float inv = 1.f / sum;

    // -------- attention-weighted aggregation + scatter --------
    float e0 = 0.f, e1 = 0.f;
#pragma unroll
    for (int k = 0; k < KNEI; ++k) {
        float a = e[k] * inv;
        f32x2 nv = *(const f32x2*)&neiS[w][k][lane * 2];
        e0 += a * nv[0];
        e1 += a * nv[1];
    }
    long un = (long)nodes[node];
    emb_out[un * EDIM + lane * 2]     = e0;
    emb_out[un * EDIM + lane * 2 + 1] = e1;
}

// ---------------------------------------------------------------------------
extern "C" void kernel_launch(void* const* d_in, const int* in_sizes, int n_in,
                              void* d_out, int out_size, void* d_ws, size_t ws_size,
                              hipStream_t stream) {
    const int*   nodes  = (const int*)  d_in[0];
    const float* n_feat = (const float*)d_in[1];
    const int*   neigh  = (const int*)  d_in[2];
    const float* u_w    = (const float*)d_in[3];
    const float* i_w    = (const float*)d_in[4];
    const float* uW     = (const float*)d_in[5];
    const float* ub     = (const float*)d_in[6];
    const float* iW     = (const float*)d_in[7];
    const float* ib     = (const float*)d_in[8];
    const float* aW1    = (const float*)d_in[9];
    const float* ab1    = (const float*)d_in[10];
    const float* aW2    = (const float*)d_in[11];
    const float* ab2    = (const float*)d_in[12];
    const float* aW3    = (const float*)d_in[13];
    const float* ab3    = (const float*)d_in[14];

    const int N  = in_sizes[0];          // 100000 nodes
    const int NU = in_sizes[3] / FDIM;   // 200000 users
    const int NI = in_sizes[4] / FDIM;   // 200000 items

    float* out_fea = (float*)d_out;                   // [N, 64]
    float* out_emb = out_fea + (size_t)N * EDIM;      // [NU, 64]
    float* iproj   = (float*)d_ws;                    // [NI, 64] (51 MB)

    // embed_matrix starts as zeros every call
    hipMemsetAsync(out_emb, 0, (size_t)NU * EDIM * sizeof(float), stream);

    // i_proj = i_weight @ iW + ib
    proj_kernel<<<(NI + 127) / 128, 128, 0, stream>>>(i_w, nullptr, iW, ib, iproj, NI);
    // nodes_fea = u_weight[nodes] @ uW + ub
    proj_kernel<<<(N + 127) / 128, 128, 0, stream>>>(u_w, nodes, uW, ub, out_fea, N);
    // attention aggregation + scatter
    attn_kernel<<<(N + 1) / 2, 64, 0, stream>>>(nodes, n_feat, neigh,
                                                aW1, ab1, aW2, ab2, aW3, ab3,
                                                iproj, out_fea, out_emb, N);
}
